// KeypointPPF_EdgeConv_5274219840018
// MI455X (gfx1250) — compile-verified, matrix operand order checked by reading, not run on hardware
//
#include <hip/hip_runtime.h>
#include <hip/hip_bf16.h>

typedef __attribute__((ext_vector_type(16))) _Float16 v16h;
typedef __attribute__((ext_vector_type(8)))  _Float16 v8h;
typedef __attribute__((ext_vector_type(8)))  float    v8f;
typedef __attribute__((ext_vector_type(4)))  unsigned int u32x4;
typedef __attribute__((ext_vector_type(8)))  int      i32x8;
typedef __attribute__((ext_vector_type(4)))  int      i32x4;

#define BB 8
#define NN 4096
#define KK 16
#define CC 128
#define EDGE_IN 384
#define BN_EPS 1e-5f

// ---------------- LDS layout (dynamic shared) ----------------
#define LDS_W_OFF    0
#define LDS_W_BYTES  (256*384*2)            // 196608 B; phase2 reuses for 256*256*2
#define LDS_P_OFF    LDS_W_BYTES
#define LDS_P_BYTES  16384
#define LDS_E_OFF    (LDS_P_OFF + LDS_P_BYTES)
#define LDS_E_STRIDE (16*EDGE_IN*2)         // 12288 B per-wave edge/h1 buffer
#define LDS_BYTES    (LDS_E_OFF + 8*LDS_E_STRIDE)   // 311296 B = 304 KB

// params region: relative byte offsets (mirrored in workspace "small" block)
#define P_E1B   0
#define P_E2B   1024
#define P_PW1   2048
#define P_PB1   2560
#define P_PW2   2688
#define P_PB2   10880
#define P_POSW  11136
#define P_POSB  11904
#define P_SMALL_BYTES 12160

// ---------------- workspace layout ----------------
#define WS_W1    0            // 256*384 f16 = 196608 B
#define WS_W2    196608       // 256*256 f16 = 131072 B
#define WS_SMALL 327680       // P_SMALL_BYTES

// ---------------- TDM availability probes ----------------
#if defined(__has_include)
#  if __has_include(<hip/amd_detail/amd_gfx1250_TDM.h>)
#    define TDM_6ARG 1
#  else
#    define TDM_6ARG 0
#  endif
#else
#  define TDM_6ARG 0
#endif

#if defined(__has_builtin)
#  if __has_builtin(__builtin_amdgcn_tensor_load_to_lds) && \
      __has_builtin(__builtin_amdgcn_s_wait_tensorcnt)
#    define USE_TDM 1
#  endif
#endif
#ifndef USE_TDM
#  define USE_TDM 0
#endif

union V16 { v16h v; v8h h[2]; };

// Load one 16-bit A/B WMMA fragment (8 dwords) from LDS: two b128 reads.
// p points at row_base + kc*32 + khalf*8 (halves); second half-group at +16.
__device__ __forceinline__ v16h ld_frag(const _Float16* p) {
    V16 r;
    r.h[0] = *(const v8h*)(p);
    r.h[1] = *(const v8h*)(p + 16);
    return r.v;
}

#if USE_TDM
// Issue one TDM copy of (rowsE8 x cols) 8-byte elements, contiguous, into LDS.
// D# per CDNA5 ISA 8.3/8.4: group0 = {count|flags, lds_addr, global_addr, type=2},
// group1 = {data_size=8B, tensor/tile dims, dim0 stride}. Groups 2/3 zero (<=2D).
__device__ __forceinline__ void tdm_load_to_lds(const void* g, unsigned ldsByteOff,
                                                unsigned rowsE8, unsigned cols) {
    unsigned long long ga = (unsigned long long)g;
    u32x4 g0;
    g0[0] = 1u;                                            // count=1, user mode
    g0[1] = ldsByteOff;                                    // lds_addr
    g0[2] = (unsigned)ga;                                  // global_addr[31:0]
    g0[3] = ((unsigned)(ga >> 32) & 0x01FFFFFFu)           // global_addr[56:32]
            | 0x80000000u;                                 // type=2 ("image")
    i32x8 g1;
    g1[0] = 0x30000;                                       // data_size=3 (8 bytes)
    g1[1] = (int)((rowsE8 & 0xFFFFu) << 16);               // tensor_dim0[15:0]
    g1[2] = (int)((cols & 0xFFFFu) << 16) | (int)(rowsE8 >> 16); // dim1 lo | dim0 hi
    g1[3] = (int)((rowsE8 & 0xFFFFu) << 16);               // tile_dim0 (dim1 hi = 0)
    g1[4] = (int)cols;                                     // tile_dim1 (tile_dim2=0)
    g1[5] = (int)rowsE8;                                   // tensor_dim0_stride[31:0]
    g1[6] = 0;                                             // stride0 hi | stride1 lo
    g1[7] = 0;
    i32x4 z4 = {};
#if TDM_6ARG
    i32x8 z8 = {};
    __builtin_amdgcn_tensor_load_to_lds(g0, g1, z4, z4, z8, 0);
#else
    __builtin_amdgcn_tensor_load_to_lds(g0, g1, z4, z4, 0);
#endif
}
#endif

// ============ prep: fold BN into e1/e2 weights, emit f16 ============
__global__ void prep_fold_big(const float* __restrict__ e1_w, const float* __restrict__ e1_g,
                              const float* __restrict__ e1_v,
                              const float* __restrict__ e2_w, const float* __restrict__ e2_g,
                              const float* __restrict__ e2_v,
                              _Float16* __restrict__ w1f, _Float16* __restrict__ w2f) {
    int i = blockIdx.x * blockDim.x + threadIdx.x;
    const int total1 = 256 * EDGE_IN;
    if (i < total1) {
        int n = i / EDGE_IN;
        float s = e1_g[n] * rsqrtf(e1_v[n] + BN_EPS);
        w1f[i] = (_Float16)(e1_w[i] * s);
    } else {
        int j = i - total1;
        if (j < 256 * 256) {
            int n = j >> 8;
            float s = e2_g[n] * rsqrtf(e2_v[n] + BN_EPS);
            w2f[j] = (_Float16)(e2_w[j] * s);
        }
    }
}

// ============ prep: fold BN into biases + small MLP params ============
__global__ void prep_small(const float* ppf_w1, const float* ppf_b1, const float* ppf_g,
                           const float* ppf_beta, const float* ppf_m, const float* ppf_v,
                           const float* ppf_w2, const float* ppf_b2,
                           const float* pos_w, const float* pos_b, const float* pos_g,
                           const float* pos_beta, const float* pos_m, const float* pos_v,
                           const float* e1_b, const float* e1_g, const float* e1_beta,
                           const float* e1_m, const float* e1_v,
                           const float* e2_b, const float* e2_g, const float* e2_beta,
                           const float* e2_m, const float* e2_v,
                           float* __restrict__ sp) {
    int t = threadIdx.x;
    if (t < 256) {
        float s1 = e1_g[t] * rsqrtf(e1_v[t] + BN_EPS);
        sp[P_E1B/4 + t] = (e1_b[t] - e1_m[t]) * s1 + e1_beta[t];
        float s2 = e2_g[t] * rsqrtf(e2_v[t] + BN_EPS);
        sp[P_E2B/4 + t] = (e2_b[t] - e2_m[t]) * s2 + e2_beta[t];
    }
    if (t < 32) {
        float s = ppf_g[t] * rsqrtf(ppf_v[t] + BN_EPS);
        sp[P_PB1/4 + t] = (ppf_b1[t] - ppf_m[t]) * s + ppf_beta[t];
        #pragma unroll
        for (int i = 0; i < 4; ++i) sp[P_PW1/4 + t*4 + i] = ppf_w1[t*4 + i] * s;
    }
    if (t < 64) {
        float s = pos_g[t] * rsqrtf(pos_v[t] + BN_EPS);
        sp[P_POSB/4 + t] = (pos_b[t] - pos_m[t]) * s + pos_beta[t];
        #pragma unroll
        for (int i = 0; i < 3; ++i) sp[P_POSW/4 + t*3 + i] = pos_w[t*3 + i] * s;
        sp[P_PB2/4 + t] = ppf_b2[t];
    }
    for (int i = t; i < 2048; i += 256) sp[P_PW2/4 + i] = ppf_w2[i];
}

// ============ main: edge build + 2x WMMA GEMM + max over K ============
__global__ __launch_bounds__(256)
void edgeconv_main(const float* __restrict__ kpt_feature, const float* __restrict__ kpt_xyz,
                   const float* __restrict__ neighbor_feature,
                   const float* __restrict__ neighbor_xyz,
                   const float* __restrict__ neighbor_normals,
                   const _Float16* __restrict__ w1f, const _Float16* __restrict__ w2f,
                   const float* __restrict__ smallp,
                   float* __restrict__ out) {
    extern __shared__ char smem[];
    _Float16* wlds = (_Float16*)(smem + LDS_W_OFF);
    float*    plds = (float*)(smem + LDS_P_OFF);

    const int tid  = threadIdx.x;
    const int wave = tid >> 5;
    const int lane = tid & 31;

    // ---- stage small params; kick off TDM DMA of folded e1 weights ----
    for (int i = tid; i < P_SMALL_BYTES/4; i += 256) plds[i] = smallp[i];
#if USE_TDM
    if (wave == 0) {
        // 196608 B = 24576 x 8B elements, as 4096 x 6 tile -> LDS offset 0
        tdm_load_to_lds(w1f, LDS_W_OFF, 4096u, 6u);
    }
    __builtin_prefetch(w2f + tid * 256, 0, 1);   // warm L2 for phase-2 weights
#else
    {
        const uint4* src = (const uint4*)w1f;
        uint4*       dst = (uint4*)wlds;
        for (int i = tid; i < (256*EDGE_IN*2)/16; i += 256) dst[i] = src[i];
        __builtin_prefetch(w2f + tid * 256, 0, 1);
    }
#endif
    __syncthreads();   // params visible to all waves (weights still in flight)

    const int g = blockIdx.x * 8 + wave;                // keypoint id = b*N + n
    _Float16* edge = (_Float16*)(smem + LDS_E_OFF + wave * LDS_E_STRIDE);

    const float* pW1  = plds + P_PW1/4;
    const float* pB1  = plds + P_PB1/4;
    const float* pW2  = plds + P_PW2/4;
    const float* pB2  = plds + P_PB2/4;
    const float* posW = plds + P_POSW/4;
    const float* posB = plds + P_POSB/4;
    const float* e1b  = plds + P_E1B/4;
    const float* e2b  = plds + P_E2B/4;

    // ------- Phase A: build 16x384 edge matrix in LDS (overlaps w1 DMA) -------
    {
        const int k    = lane >> 1;        // 2 lanes per neighbor
        const int half = lane & 1;
        const int base = (g * KK + k) * 3;
        float nx0 = neighbor_xyz[base+0], nx1 = neighbor_xyz[base+1], nx2 = neighbor_xyz[base+2];
        float nn0 = neighbor_normals[base+0], nn1 = neighbor_normals[base+1], nn2 = neighbor_normals[base+2];
        float kx0 = kpt_xyz[g*3+0], kx1 = kpt_xyz[g*3+1], kx2 = kpt_xyz[g*3+2];

        // kn = normalize(mean_k normals): each k counted twice, scale cancels
        float s0 = nn0, s1 = nn1, s2 = nn2;
        #pragma unroll
        for (int m = 1; m <= 16; m <<= 1) {
            s0 += __shfl_xor(s0, m); s1 += __shfl_xor(s1, m); s2 += __shfl_xor(s2, m);
        }
        float kl = fmaxf(sqrtf(s0*s0 + s1*s1 + s2*s2), 1e-12f);
        float kn0 = s0/kl, kn1 = s1/kl, kn2 = s2/kl;

        float r0 = nx0-kx0, r1 = nx1-kx1, r2 = nx2-kx2;
        float dn = sqrtf(r0*r0 + r1*r1 + r2*r2);
        float iv = 1.0f / (dn + 1e-8f);
        float d0 = r0*iv, d1 = r1*iv, d2 = r2*iv;
        float al = fminf(fmaxf(kn0*d0 + kn1*d1 + kn2*d2, -1.f), 1.f);
        float ph = fminf(fmaxf(nn0*d0 + nn1*d1 + nn2*d2, -1.f), 1.f);
        float th = fminf(fmaxf(kn0*nn0 + kn1*nn1 + kn2*nn2, -1.f), 1.f);

        float h32v[32];
        #pragma unroll
        for (int j = 0; j < 32; ++j) {
            float a = dn*pW1[j*4+0] + al*pW1[j*4+1] + ph*pW1[j*4+2] + th*pW1[j*4+3] + pB1[j];
            h32v[j] = fmaxf(a, 0.f);
        }
        const int ob = half * 32;                     // each lane: 32 output chans
        _Float16* erow = edge + k * EDGE_IN;
        for (int o = 0; o < 32; ++o) {
            int oo = ob + o;
            float acc = pB2[oo];
            #pragma unroll
            for (int j = 0; j < 32; ++j) acc += h32v[j] * pW2[oo*32 + j];
            erow[256 + oo] = (_Float16)acc;           // ppf_enc (no relu layer 2)
            float p = r0*posW[oo*3+0] + r1*posW[oo*3+1] + r2*posW[oo*3+2] + posB[oo];
            erow[320 + oo] = (_Float16)fmaxf(p, 0.f); // pos_enc
        }
    }
    {   // center + diff (all 32 lanes)
        const float* cf = kpt_feature      + g * CC;
        const float* nf = neighbor_feature + g * KK * CC;
        for (int t = lane; t < KK*CC; t += 32) {
            int k = t >> 7, c = t & 127;
            float cv = cf[c];
            float dv = nf[t] - cv;
            edge[k*EDGE_IN + c]      = (_Float16)cv;
            edge[k*EDGE_IN + CC + c] = (_Float16)dv;
        }
    }
#if USE_TDM
    if (wave == 0) __builtin_amdgcn_s_wait_tensorcnt(0);  // w1 DMA complete
#endif
    __syncthreads();

    // ------- GEMM1: (16x384) x (384x256), f16 WMMA, f32 accum -------
    const int mrow  = lane & 15;   // A: M row / B: N col owned by this lane
    const int khalf = lane >> 4;   // which K half-group this lane carries
    v8f zacc = {};
    v8f acc[16];
    #pragma unroll
    for (int nt = 0; nt < 16; ++nt) acc[nt] = zacc;

    for (int kc = 0; kc < EDGE_IN/32; ++kc) {
        v16h a = ld_frag(edge + mrow*EDGE_IN + kc*32 + khalf*8);
        #pragma unroll
        for (int nt = 0; nt < 16; ++nt) {
            v16h bfr = ld_frag(wlds + (nt*16 + mrow)*EDGE_IN + kc*32 + khalf*8);
            acc[nt] = __builtin_amdgcn_wmma_f32_16x16x32_f16(
                false, a, false, bfr, (short)0, acc[nt], false, false);
        }
    }
    __syncthreads();   // all waves done reading e1 weights from LDS

#if USE_TDM
    if (wave == 0) {
        // 131072 B = 16384 x 8B elements, as 4096 x 4 tile -> overwrite wlds
        tdm_load_to_lds(w2f, LDS_W_OFF, 4096u, 4u);   // overlaps epilogue below
    }
#endif

    // epilogue 1: bias + relu -> h1 (f16) in per-wave LDS (reuse edge buffer)
    _Float16* h1 = edge;   // 16 x 256, row stride 256 halves
    #pragma unroll
    for (int nt = 0; nt < 16; ++nt) {
        float bias = e1b[nt*16 + mrow];
        #pragma unroll
        for (int r = 0; r < 8; ++r) {
            float v = fmaxf(acc[nt][r] + bias, 0.f);
            int m = r + khalf*8;
            h1[m*256 + nt*16 + mrow] = (_Float16)v;
        }
    }

#if USE_TDM
    if (wave == 0) __builtin_amdgcn_s_wait_tensorcnt(0);  // w2 DMA complete
#else
    {
        const uint4* src = (const uint4*)w2f;
        uint4*       dst = (uint4*)wlds;
        for (int i = tid; i < (256*256*2)/16; i += 256) dst[i] = src[i];
    }
#endif
    __syncthreads();

    // ------- GEMM2: (16x256) x (256x256) -------
    #pragma unroll
    for (int nt = 0; nt < 16; ++nt) acc[nt] = zacc;

    for (int kc = 0; kc < 256/32; ++kc) {
        v16h a = ld_frag(h1 + mrow*256 + kc*32 + khalf*8);
        #pragma unroll
        for (int nt = 0; nt < 16; ++nt) {
            v16h bfr = ld_frag(wlds + (nt*16 + mrow)*256 + kc*32 + khalf*8);
            acc[nt] = __builtin_amdgcn_wmma_f32_16x16x32_f16(
                false, a, false, bfr, (short)0, acc[nt], false, false);
        }
    }

    // epilogue 2: bias, max over K rows (relu folds into final max), store f32
    float* orow = out + g * 256;
    #pragma unroll
    for (int nt = 0; nt < 16; ++nt) {
        float bias = e2b[nt*16 + mrow];
        float mx = -3.4e38f;
        #pragma unroll
        for (int r = 0; r < 8; ++r) mx = fmaxf(mx, acc[nt][r] + bias);
        mx = fmaxf(mx, __shfl_xor(mx, 16));       // combine the two M half-groups
        if (khalf == 0) orow[nt*16 + mrow] = fmaxf(mx, 0.f);
    }
}

extern "C" void kernel_launch(void* const* d_in, const int* in_sizes, int n_in,
                              void* d_out, int out_size, void* d_ws, size_t ws_size,
                              hipStream_t stream) {
    (void)in_sizes; (void)n_in; (void)out_size; (void)ws_size;
    const float* kpt_feature      = (const float*)d_in[0];
    const float* kpt_xyz          = (const float*)d_in[1];
    const float* neighbor_feature = (const float*)d_in[2];
    const float* neighbor_xyz     = (const float*)d_in[3];
    const float* neighbor_normals = (const float*)d_in[4];
    const float* ppf_w1  = (const float*)d_in[5];
    const float* ppf_b1  = (const float*)d_in[6];
    const float* ppf_g   = (const float*)d_in[7];
    const float* ppf_beta= (const float*)d_in[8];
    const float* ppf_m   = (const float*)d_in[9];
    const float* ppf_v   = (const float*)d_in[10];
    const float* ppf_w2  = (const float*)d_in[11];
    const float* ppf_b2  = (const float*)d_in[12];
    const float* pos_w   = (const float*)d_in[13];
    const float* pos_b   = (const float*)d_in[14];
    const float* pos_g   = (const float*)d_in[15];
    const float* pos_beta= (const float*)d_in[16];
    const float* pos_m   = (const float*)d_in[17];
    const float* pos_v   = (const float*)d_in[18];
    const float* e1_w    = (const float*)d_in[19];
    const float* e1_b    = (const float*)d_in[20];
    const float* e1_g    = (const float*)d_in[21];
    const float* e1_beta = (const float*)d_in[22];
    const float* e1_m    = (const float*)d_in[23];
    const float* e1_v    = (const float*)d_in[24];
    const float* e2_w    = (const float*)d_in[25];
    const float* e2_b    = (const float*)d_in[26];
    const float* e2_g    = (const float*)d_in[27];
    const float* e2_beta = (const float*)d_in[28];
    const float* e2_m    = (const float*)d_in[29];
    const float* e2_v    = (const float*)d_in[30];

    char* ws = (char*)d_ws;
    _Float16* w1f = (_Float16*)(ws + WS_W1);
    _Float16* w2f = (_Float16*)(ws + WS_W2);
    float*  smallp = (float*)(ws + WS_SMALL);

    // fold BN into weights (f16) and biases
    prep_fold_big<<<(256*EDGE_IN + 256*256 + 255)/256, 256, 0, stream>>>(
        e1_w, e1_g, e1_v, e2_w, e2_g, e2_v, w1f, w2f);
    prep_small<<<1, 256, 0, stream>>>(
        ppf_w1, ppf_b1, ppf_g, ppf_beta, ppf_m, ppf_v, ppf_w2, ppf_b2,
        pos_w, pos_b, pos_g, pos_beta, pos_m, pos_v,
        e1_b, e1_g, e1_beta, e1_m, e1_v,
        e2_b, e2_g, e2_beta, e2_m, e2_v, smallp);

    // main fused kernel: 8 keypoints per block (1 per wave), 4096 blocks
    edgeconv_main<<<(BB*NN)/8, 256, LDS_BYTES, stream>>>(
        kpt_feature, kpt_xyz, neighbor_feature, neighbor_xyz, neighbor_normals,
        w1f, w2f, smallp, (float*)d_out);
}
